// PGWCrossAttention_9835475107931
// MI455X (gfx1250) — matrix operand order repaired and checked by reading, compiled, not run on hardware
//
#include <hip/hip_runtime.h>
#include <hip/hip_bf16.h>

typedef __attribute__((ext_vector_type(16))) _Float16     v16h;
typedef __attribute__((ext_vector_type(8)))  _Float16     v8h;
typedef __attribute__((ext_vector_type(8)))  float        v8f;
typedef __attribute__((ext_vector_type(4)))  unsigned int v4u;
typedef __attribute__((ext_vector_type(8)))  int          v8i;
typedef __attribute__((ext_vector_type(4)))  int          v4i;

#define WMMA_F16(A, B, C) \
  __builtin_amdgcn_wmma_f32_16x16x32_f16(false, (A), false, (B), (short)0, (C), false, false)

static constexpr int Bsz  = 16;
static constexpr int LQ   = 2048;
static constexpr int LK   = 2048;
static constexpr int DIN  = 1024;
static constexpr int H    = 256;
static constexpr float LOG2E = 1.44269504088896340736f;

// ---------------------------------------------------------------------------
// TDM: 2D tile load Global -> LDS (D# per CDNA5 ISA).
// data_size=1 (2-byte elements). Groups 2/3 zero (<=2D tensor). Issued per
// wave (EXEC ignored) -> callers gate on wave 0. Tracked by TENSORcnt.
// ---------------------------------------------------------------------------
__device__ __forceinline__ void tdm_load_2d(const void* gaddr, unsigned lds_addr,
                                            unsigned dim0, unsigned dim1,
                                            unsigned stride0,
                                            unsigned tile0, unsigned tile1) {
  unsigned long long ga = (unsigned long long)(uintptr_t)gaddr;
  v4u g0;
  g0.x = 1u;                                               // count=1, user mode
  g0.y = lds_addr;                                         // LDS byte address
  g0.z = (unsigned)(ga & 0xffffffffu);                     // global_addr[31:0]
  g0.w = (unsigned)((ga >> 32) & 0x01ffffffu) | (2u << 30); // addr[56:32] | type=2
  v8i g1;
  g1[0] = (int)(1u << 16);                                 // data_size=1 (2B); mask=0
  g1[1] = (int)((dim0 & 0xffffu) << 16);                   // tensor_dim0[15:0]
  g1[2] = (int)(((dim0 >> 16) & 0xffffu) | ((dim1 & 0xffffu) << 16));
  g1[3] = (int)(((dim1 >> 16) & 0xffffu) | ((tile0 & 0xffffu) << 16));
  g1[4] = (int)(tile1 & 0xffffu);                          // tile_dim1 (tile_dim2=0)
  g1[5] = (int)stride0;                                    // tensor_dim0_stride[31:0]
  g1[6] = 0;
  g1[7] = 0;
  v4i z4 = {0, 0, 0, 0};
#if __clang_major__ >= 23
  v8i z8 = {0, 0, 0, 0, 0, 0, 0, 0};
  __builtin_amdgcn_tensor_load_to_lds(g0, g1, z4, z4, z8, 0);
#else
  __builtin_amdgcn_tensor_load_to_lds(g0, g1, z4, z4, 0);
#endif
}

// ---------------------------------------------------------------------------
// Weight transpose + f32->f16: Wt[n][k] = (f16)W[k][n], W is [DIN][H].
// ---------------------------------------------------------------------------
__global__ void wt_transpose_kernel(const float* __restrict__ W,
                                    _Float16* __restrict__ Wt) {
  int idx = blockIdx.x * 256 + threadIdx.x;
  int k = idx / H;
  int n = idx % H;
  Wt[(size_t)n * DIN + k] = (_Float16)W[idx];
}

// ---------------------------------------------------------------------------
// Projection GEMM: Out = (A[M x DIN] @ W + bias) * scale, f16 output.
// Software-pipelined k-loop (next-stage fragments loaded before WMMAs).
// ---------------------------------------------------------------------------
template <bool TRANSPOSED>
__global__ __launch_bounds__(256)
void proj_kernel(const float* __restrict__ A, const _Float16* __restrict__ Wt,
                 const float* __restrict__ bias, _Float16* __restrict__ Out,
                 float scale) {
  const int w    = threadIdx.x >> 5;
  const int lane = threadIdx.x & 31;
  const int ln15 = lane & 15;
  const int hg   = lane >> 4;

  const int row0 = blockIdx.x * 128 + w * 16;
  const int n0   = blockIdx.y * 32;

  const float*    arow = A  + (size_t)(row0 + ln15) * DIN;
  const _Float16* w0   = Wt + (size_t)(n0      + ln15) * DIN + 16 * hg;
  const _Float16* w1   = Wt + (size_t)(n0 + 16 + ln15) * DIN + 16 * hg;
  const int cA = 16 * hg;

  auto loadfrag = [&](int k0, v16h& af, v16h& bb0, v16h& bb1) {
    v8f lo = *(const v8f*)(arow + k0 + cA);
    v8f hi = *(const v8f*)(arow + k0 + cA + 8);
#pragma unroll
    for (int i = 0; i < 8; i++) { af[i] = (_Float16)lo[i]; af[i + 8] = (_Float16)hi[i]; }
    bb0 = *(const v16h*)(w0 + k0);
    bb1 = *(const v16h*)(w1 + k0);
  };

  v8f acc0 = {}, acc1 = {};
  v16h af, b0, b1;
  loadfrag(0, af, b0, b1);
  for (int k0 = 32; k0 < DIN; k0 += 32) {
    v16h afn, b0n, b1n;
    loadfrag(k0, afn, b0n, b1n);
    acc0 = WMMA_F16(af, b0, acc0);
    acc1 = WMMA_F16(af, b1, acc1);
    af = afn; b0 = b0n; b1 = b1n;
  }
  acc0 = WMMA_F16(af, b0, acc0);
  acc1 = WMMA_F16(af, b1, acc1);

  const float bia0 = bias[n0 + ln15];
  const float bia1 = bias[n0 + 16 + ln15];

  if constexpr (!TRANSPOSED) {
#pragma unroll
    for (int r = 0; r < 8; r++) {
      const int m = r + 8 * hg;
      const size_t o = (size_t)(row0 + m) * H;
      Out[o + n0 + ln15]      = (_Float16)((acc0[r] + bia0) * scale);
      Out[o + n0 + 16 + ln15] = (_Float16)((acc1[r] + bia1) * scale);
    }
  } else {
    v8h p0, p1;
#pragma unroll
    for (int r = 0; r < 8; r++) {
      p0[r] = (_Float16)((acc0[r] + bia0) * scale);
      p1[r] = (_Float16)((acc1[r] + bia1) * scale);
    }
    const int bb  = row0 >> 11;
    const int lk0 = (row0 & (LK - 1)) + 8 * hg;
    *(v8h*)(Out + ((size_t)bb * H + n0      + ln15) * LK + lk0) = p0;
    *(v8h*)(Out + ((size_t)bb * H + n0 + 16 + ln15) * LK + lk0) = p1;
  }
}

// ---------------------------------------------------------------------------
// Fused flash attention, TDM-staged operands. 4 waves/block, one 16-row q
// tile per wave; all waves share the same K/V tiles.
//   - K tile (32x256 f16, 16KB) double-buffered in LDS, TDM-prefetched one
//     step ahead by wave 0 (TENSORcnt + barrier handoff).
//   - V tile (256x32 f16, 16KB) single-buffered: issued at step top, needed
//     only after S + softmax -> DMA hidden behind 16 WMMAs + VALU.
//   - pi_star: NT global loads issued before the WMMAs (dominant HBM stream).
//   - Q A-fragments register-resident for the whole loop (64 VGPRs).
// ---------------------------------------------------------------------------
__global__ __launch_bounds__(128)
void attn_kernel(const _Float16* __restrict__ Q, const _Float16* __restrict__ K,
                 const _Float16* __restrict__ Vt, const float* __restrict__ pi,
                 const float* __restrict__ gammaPtr, float* __restrict__ out) {
  __shared__ __attribute__((aligned(32))) _Float16 ldsK[2][32 * 256]; // 32KB dbl K
  __shared__ __attribute__((aligned(32))) _Float16 ldsV[256 * 32];    // 16KB V
  __shared__ __attribute__((aligned(32))) _Float16 ldsP[4][16 * 32];  // 4KB  P bounce

  const int w    = threadIdx.x >> 5;
  const int lane = threadIdx.x & 31;
  const int ln15 = lane & 15;
  const int hg   = lane >> 4;

  const int b  = blockIdx.y;
  const int q0 = blockIdx.x * 64 + w * 16;

  const _Float16* Qb  = Q  + ((size_t)b * LQ + q0) * H;
  const _Float16* Kb  = K  + (size_t)b * LK * H;
  const _Float16* Vb  = Vt + (size_t)b * H * LK;
  const float*    pib = pi + (size_t)b * LQ * LK + (size_t)q0 * LK;
  const float gamma = gammaPtr[0];

  // ---- Q A-fragments: register-resident for the whole k-loop ----
  v16h qf[8];
#pragma unroll
  for (int hs = 0; hs < 8; hs++) {
    const _Float16* qp = Qb + (size_t)ln15 * H + hs * 32 + 16 * hg;
    v8h lo = *(const v8h*)qp;
    v8h hi = *(const v8h*)(qp + 8);
#pragma unroll
    for (int i = 0; i < 8; i++) { qf[hs][i] = lo[i]; qf[hs][i + 8] = hi[i]; }
  }

  // ---- TDM prologue: stage first K tile ----
  if (w == 0) {
    tdm_load_2d(Kb, (unsigned)(uintptr_t)&ldsK[0][0], H, LK, H, H, 32);
    __builtin_amdgcn_s_wait_tensorcnt(0);
  }
  __syncthreads();

  v8f o[16] = {};
  float m[8], l[8];
#pragma unroll
  for (int r = 0; r < 8; r++) { m[r] = -3.0e38f; l[r] = 0.0f; }

  _Float16* myP = &ldsP[w][0];

  for (int k0 = 0; k0 < LK; k0 += 32) {
    const int cur = (k0 >> 5) & 1;
    const _Float16* Kcur = &ldsK[cur][0];

    // ---- TDM: prefetch next K tile + this step's V tile (wave 0 only) ----
    if (w == 0) {
      if (k0 + 32 < LK)
        tdm_load_2d(Kb + (size_t)(k0 + 32) * H,
                    (unsigned)(uintptr_t)&ldsK[cur ^ 1][0], H, LK, H, H, 32);
      tdm_load_2d(Vb + k0, (unsigned)(uintptr_t)&ldsV[0], LK, H, LK, 32, H);
    }

    // ---- pi_star: NT loads issued before all WMMAs ----
    float pv0[8], pv1[8];
#pragma unroll
    for (int r = 0; r < 8; r++) {
      const size_t prow = (size_t)(r + 8 * hg) * LK + k0;
      pv0[r] = __builtin_nontemporal_load(pib + prow + ln15);
      pv1[r] = __builtin_nontemporal_load(pib + prow + 16 + ln15);
    }

    // ---- S = Qs @ K^T : 8 h-steps of K=32, B fragments from LDS ----
    auto loadS = [&](int h0, v16h& bb0, v16h& bb1) {
      bb0 = *(const v16h*)(Kcur + ln15 * 256 + h0 + 16 * hg);
      bb1 = *(const v16h*)(Kcur + (16 + ln15) * 256 + h0 + 16 * hg);
    };
    v8f s0 = {}, s1 = {};
    v16h kb0, kb1;
    loadS(0, kb0, kb1);
#pragma unroll
    for (int hs = 0; hs < 8; hs++) {
      v16h b0n, b1n;
      if (hs < 7) loadS((hs + 1) * 32, b0n, b1n);
      s0 = WMMA_F16(qf[hs], kb0, s0);
      s1 = WMMA_F16(qf[hs], kb1, s1);
      if (hs < 7) { kb0 = b0n; kb1 = b1n; }
    }

    // ---- add log-prior bias: gamma * log2(clip(pi,1e-9)) (base-2 domain) ----
#pragma unroll
    for (int r = 0; r < 8; r++) {
      s0[r] += gamma * __builtin_amdgcn_logf(fmaxf(pv0[r], 1e-9f));
      s1[r] += gamma * __builtin_amdgcn_logf(fmaxf(pv1[r], 1e-9f));
    }

    // ---- online softmax (base-2); rows in (r,hg), N across 16 lanes ----
#pragma unroll
    for (int r = 0; r < 8; r++) {
      float mx = fmaxf(s0[r], s1[r]);
      mx = fmaxf(mx, __shfl_xor(mx, 1, 32));
      mx = fmaxf(mx, __shfl_xor(mx, 2, 32));
      mx = fmaxf(mx, __shfl_xor(mx, 4, 32));
      mx = fmaxf(mx, __shfl_xor(mx, 8, 32));
      const float mn = fmaxf(m[r], mx);
      const float sc = __builtin_amdgcn_exp2f(m[r] - mn);
      m[r] = mn;
      const float p0 = __builtin_amdgcn_exp2f(s0[r] - mn);
      const float p1 = __builtin_amdgcn_exp2f(s1[r] - mn);
      s0[r] = p0; s1[r] = p1;
      float rs = p0 + p1;
      rs += __shfl_xor(rs, 1, 32);
      rs += __shfl_xor(rs, 2, 32);
      rs += __shfl_xor(rs, 4, 32);
      rs += __shfl_xor(rs, 8, 32);
      l[r] = l[r] * sc + rs;
#pragma unroll
      for (int t = 0; t < 16; t++) o[t][r] *= sc;
    }

    // ---- P: C-layout -> A-layout via wave-local LDS bounce (f32 -> f16) ----
#pragma unroll
    for (int r = 0; r < 8; r++) {
      const int mrow = r + 8 * hg;
      myP[mrow * 32 + ln15]      = (_Float16)s0[r];
      myP[mrow * 32 + 16 + ln15] = (_Float16)s1[r];
    }
    const _Float16* pp = myP + ln15 * 32 + 16 * hg;
    v8h plo = *(const v8h*)pp;
    v8h phi = *(const v8h*)(pp + 8);
    v16h pa;
#pragma unroll
    for (int i = 0; i < 8; i++) { pa[i] = plo[i]; pa[i + 8] = phi[i]; }

    // ---- handoff: V tile (and next K tile) landed; publish to all waves ----
    if (w == 0) __builtin_amdgcn_s_wait_tensorcnt(0);
    __syncthreads();

    // ---- O += P @ V : 16 h-tiles, B fragments from LDS V tile ----
    const _Float16* vb = &ldsV[0] + ln15 * 32 + 16 * hg;
    v16h bv = *(const v16h*)(vb);
#pragma unroll
    for (int t = 0; t < 16; t++) {
      v16h bvn;
      if (t < 15) bvn = *(const v16h*)(vb + (t + 1) * 16 * 32);
      o[t] = WMMA_F16(pa, bv, o[t]);
      if (t < 15) bv = bvn;
    }

    // ---- all waves done with Kcur / ldsV before next step's TDM issues ----
    __syncthreads();
  }

  // ---- epilogue: O / l, streamed out with NT stores ----
  float inv[8];
#pragma unroll
  for (int r = 0; r < 8; r++) inv[r] = 1.0f / l[r];
  float* ob = out + ((size_t)b * LQ + q0) * H;
#pragma unroll
  for (int t = 0; t < 16; t++) {
#pragma unroll
    for (int r = 0; r < 8; r++) {
      __builtin_nontemporal_store(o[t][r] * inv[r],
                                  ob + (size_t)(r + 8 * hg) * H + t * 16 + ln15);
    }
  }
}

// ---------------------------------------------------------------------------
extern "C" void kernel_launch(void* const* d_in, const int* in_sizes, int n_in,
                              void* d_out, int out_size, void* d_ws, size_t ws_size,
                              hipStream_t stream) {
  const float* q_fp  = (const float*)d_in[0];
  const float* v_ret = (const float*)d_in[1];
  const float* pi    = (const float*)d_in[2];
  const float* Wq    = (const float*)d_in[3];
  const float* bq    = (const float*)d_in[4];
  const float* Wk    = (const float*)d_in[5];
  const float* bk    = (const float*)d_in[6];
  const float* Wv    = (const float*)d_in[7];
  const float* bv    = (const float*)d_in[8];
  const float* gamma = (const float*)d_in[9];
  float* outp        = (float*)d_out;

  _Float16* Wtq = (_Float16*)d_ws;
  _Float16* Wtk = Wtq + (size_t)H * DIN;
  _Float16* Wtv = Wtk + (size_t)H * DIN;
  _Float16* Qh  = Wtv + (size_t)H * DIN;
  _Float16* Kh  = Qh  + (size_t)Bsz * LQ * H;
  _Float16* Vth = Kh  + (size_t)Bsz * LK * H;

  wt_transpose_kernel<<<(DIN * H) / 256, 256, 0, stream>>>(Wq, Wtq);
  wt_transpose_kernel<<<(DIN * H) / 256, 256, 0, stream>>>(Wk, Wtk);
  wt_transpose_kernel<<<(DIN * H) / 256, 256, 0, stream>>>(Wv, Wtv);

  const dim3 pgrid(256, 8);
  const float qscale = LOG2E / 16.0f;   // log2e / sqrt(H), folded into Q
  proj_kernel<false><<<pgrid, 256, 0, stream>>>(q_fp,  Wtq, bq, Qh,  qscale);
  proj_kernel<false><<<pgrid, 256, 0, stream>>>(v_ret, Wtk, bk, Kh,  1.0f);
  proj_kernel<true ><<<pgrid, 256, 0, stream>>>(v_ret, Wtv, bv, Vth, 1.0f);

  attn_kernel<<<dim3(LQ / 64, Bsz), 128, 0, stream>>>(Qh, Kh, Vth, pi, gamma, outp);

  (void)in_sizes; (void)n_in; (void)out_size; (void)ws_size;
}
